// ImportanceRenderer_14998025797801
// MI455X (gfx1250) — matrix operand order repaired and checked by reading, not compiled, not run on hardware
//
#include <hip/hip_runtime.h>

// Problem constants (from reference setup_inputs)
#define B_   2
#define R_   4096
#define N_   48
#define F_   32
#define HP_  256
#define HID_ 64
#define PPB  (R_ * N_)      // points per batch = 196608
#define BP   (B_ * PPB)     // total points     = 393216
#define NTILES (BP / 16)    // 16-point WMMA tiles = 24576

typedef __attribute__((ext_vector_type(16))) _Float16 v16h;
typedef __attribute__((ext_vector_type(8)))  _Float16 v8h;
typedef __attribute__((ext_vector_type(8)))  float    v8f;

#define LOG2E_F 1.4426950408889634f
#define LN2_F   0.6931471805599453f

// Raw-hardware-transcendental softplus: max(x,0) + ln2*log2(1 + 2^(-|x|*log2e)).
// Argument of log2 is in [1,2]: no denormal/special-case fixups needed, so we
// use bare v_exp_f32 / v_log_f32 via the amdgcn builtins (branch-free,
// co-executes with the XDL WMMA pipe).
__device__ __forceinline__ float softplus_f(float x) {
  float e = __builtin_amdgcn_exp2f(-fabsf(x) * LOG2E_F);
  float l = __builtin_amdgcn_logf(1.f + e);        // log2(1+e), e in (0,1]
  return fmaxf(x, 0.f) + l * LN2_F;
}
__device__ __forceinline__ float sigmoid_f(float x) {
  float e = __builtin_amdgcn_exp2f(-x * LOG2E_F);
  return __builtin_amdgcn_rcpf(1.f + e);
}

// ---------------------------------------------------------------------------
// Kernel 1: tri-plane bilinear sampling -> f16 feature rows [BP][32]
// ---------------------------------------------------------------------------
__global__ void feat_kernel(const float* __restrict__ planes,
                            const float* __restrict__ ro,
                            const float* __restrict__ rd,
                            const float* __restrict__ jit,
                            _Float16* __restrict__ xfeat) {
  int p = blockIdx.x * blockDim.x + threadIdx.x;
  if (p >= BP) return;
  int b   = p / PPB;
  int pp  = p - b * PPB;
  int ray = pp / N_;
  int n   = pp - ray * N_;

  const float sdel = 1.f / (N_ - 1);
  float sd = (float)n * sdel + jit[(size_t)(b * R_ + ray) * N_ + n] * sdel;
  float t  = sd;                       // RAY_START=0, RAY_END=1
  size_t rb = (size_t)(b * R_ + ray) * 3;
  float c[3];
  c[0] = ro[rb + 0] + t * rd[rb + 0];  // BOX_SIZE/2 == 1 -> no divide
  c[1] = ro[rb + 1] + t * rd[rb + 1];
  c[2] = ro[rb + 2] + t * rd[rb + 2];

  float w4[3][4];
  int   idx4[3][4];
#pragma unroll
  for (int pl = 0; pl < 3; ++pl) {
    // plane uv pairs: (c0,c1), (c0,c2), (c1,c2)
    float gx = (c[(pl == 2) ? 1 : 0] + 1.f) * 0.5f * (HP_ - 1);
    float gy = (c[(pl == 0) ? 1 : 2] + 1.f) * 0.5f * (HP_ - 1);
    float fx0 = floorf(gx), fy0 = floorf(gy);
    int x0 = (int)fx0, y0 = (int)fy0;
    int x1 = x0 + 1,  y1 = y0 + 1;
    float wx = gx - fx0, wy = gy - fy0;
    bool vx0 = (x0 >= 0) && (x0 < HP_), vx1 = (x1 >= 0) && (x1 < HP_);
    bool vy0 = (y0 >= 0) && (y0 < HP_), vy1 = (y1 >= 0) && (y1 < HP_);
    int cx0 = min(max(x0, 0), HP_ - 1), cx1 = min(max(x1, 0), HP_ - 1);
    int cy0 = min(max(y0, 0), HP_ - 1), cy1 = min(max(y1, 0), HP_ - 1);
    w4[pl][0] = (vx0 && vy0) ? (1.f - wx) * (1.f - wy) : 0.f;
    w4[pl][1] = (vx1 && vy0) ? wx * (1.f - wy) : 0.f;
    w4[pl][2] = (vx0 && vy1) ? (1.f - wx) * wy : 0.f;
    w4[pl][3] = (vx1 && vy1) ? wx * wy : 0.f;
    idx4[pl][0] = cy0 * HP_ + cx0;
    idx4[pl][1] = cy0 * HP_ + cx1;
    idx4[pl][2] = cy1 * HP_ + cx0;
    idx4[pl][3] = cy1 * HP_ + cx1;
  }

  size_t pb = (size_t)b * 3 * F_ * HP_ * HP_;
  for (int f = 0; f < F_; ++f) {
    float acc = 0.f;
#pragma unroll
    for (int pl = 0; pl < 3; ++pl) {
      const float* img = planes + pb + ((size_t)pl * F_ + f) * (HP_ * HP_);
      acc += img[idx4[pl][0]] * w4[pl][0] + img[idx4[pl][1]] * w4[pl][1] +
             img[idx4[pl][2]] * w4[pl][2] + img[idx4[pl][3]] * w4[pl][3];
    }
    xfeat[(size_t)p * F_ + f] = (_Float16)(acc * (1.f / 3.f));
  }
}

// ---------------------------------------------------------------------------
// Kernel 2: fused 2-layer MLP via v_wmma_f32_16x16x32_f16.
// One wave per 16-point tile (grid-stride). Layer-1 h bounced through
// per-wave LDS to convert D-fragment -> A-fragment layout.
// ---------------------------------------------------------------------------
__global__ void mlp_kernel(const _Float16* __restrict__ xfeat,  // [BP][32]
                           const float* __restrict__ W1,        // [32][64]
                           const float* __restrict__ b1,        // [64]
                           const float* __restrict__ W2,        // [64][4]
                           const float* __restrict__ b2,        // [4]
                           float* __restrict__ oBuf) {          // [BP][4]
  __shared__ _Float16 lds_h[8][16 * HID_];   // 16KB: per-wave h staging
  const int lane   = threadIdx.x & 31;
  const int waveIb = threadIdx.x >> 5;
  const int col    = lane & 15;   // N index (and A-row index for this lane)
  const int g      = lane >> 4;   // half-wave group

  // ---- W1 as 4 B-fragments (N-tiles of 16), f32 -> f16 on load ----
  v16h bW1[4];
#pragma unroll
  for (int nt = 0; nt < 4; ++nt)
#pragma unroll
    for (int e = 0; e < 16; ++e)
      bW1[nt][e] = (_Float16)W1[(g * 8 + (e & 7) + (e >> 3) * 16) * HID_ + nt * 16 + col];

  // ---- W2 as 2 K-chunk B-fragments, N padded 4 -> 16 with zeros ----
  v16h bW2[2];
#pragma unroll
  for (int kc = 0; kc < 2; ++kc)
#pragma unroll
    for (int e = 0; e < 16; ++e)
      bW2[kc][e] = (col < 4) ? (_Float16)W2[(kc * 32 + g * 8 + (e & 7) + (e >> 3) * 16) * 4 + col]
                             : (_Float16)0.0f;

  float b1v[4];
#pragma unroll
  for (int nt = 0; nt < 4; ++nt) b1v[nt] = b1[nt * 16 + col];
  float b2v = (col < 4) ? b2[col] : 0.f;

  _Float16* hrow = &lds_h[waveIb][0];
  const int wavesTotal = gridDim.x * (blockDim.x >> 5);
  int wave = blockIdx.x * (blockDim.x >> 5) + waveIb;

  for (int tile = wave; tile < NTILES; tile += wavesTotal) {
    const size_t base = (size_t)tile * 16;
    const int row = col;  // A-matrix row held by this lane

    // A fragment of x: lane holds K [8g,8g+8) and [16+8g,16+8g+8) of its row
    const _Float16* xr = xfeat + (base + row) * F_;
    v8h lo = *(const v8h*)(xr + g * 8);
    v8h hi = *(const v8h*)(xr + 16 + g * 8);
    v16h a;
#pragma unroll
    for (int e = 0; e < 8; ++e) { a[e] = lo[e]; a[8 + e] = hi[e]; }

    // ---- layer 1: x[16x32] @ W1[32x64], bias in C ----
    v8f acc[4];
#pragma unroll
    for (int nt = 0; nt < 4; ++nt) {
#pragma unroll
      for (int r2 = 0; r2 < 8; ++r2) acc[nt][r2] = b1v[nt];
      acc[nt] = __builtin_amdgcn_wmma_f32_16x16x32_f16(
          false, a, false, bW1[nt], (short)0, acc[nt], false, false);
    }

    // softplus (raw v_exp/v_log, branch-free), write h to LDS
    // (D layout: lane holds rows r2+8g, col nt*16+col)
#pragma unroll
    for (int nt = 0; nt < 4; ++nt)
#pragma unroll
      for (int r2 = 0; r2 < 8; ++r2)
        hrow[(r2 + 8 * g) * HID_ + nt * 16 + col] =
            (_Float16)softplus_f(acc[nt][r2]);
    // same-wave DS ops are in-order: loads below see the stores above

    // re-gather h as A-fragments for the two K=32 chunks
    v16h ah0, ah1;
#pragma unroll
    for (int e = 0; e < 8; ++e) {
      ah0[e]     = hrow[row * HID_ + g * 8 + e];
      ah0[8 + e] = hrow[row * HID_ + 16 + g * 8 + e];
      ah1[e]     = hrow[row * HID_ + 32 + g * 8 + e];
      ah1[8 + e] = hrow[row * HID_ + 32 + 16 + g * 8 + e];
    }

    // ---- layer 2: h[16x64] @ W2[64x4 padded to 16] ----
    v8f oacc;
#pragma unroll
    for (int r2 = 0; r2 < 8; ++r2) oacc[r2] = b2v;
    oacc = __builtin_amdgcn_wmma_f32_16x16x32_f16(
        false, ah0, false, bW2[0], (short)0, oacc, false, false);
    oacc = __builtin_amdgcn_wmma_f32_16x16x32_f16(
        false, ah1, false, bW2[1], (short)0, oacc, false, false);

    // write o logits [point][4]; only N-cols 0..3 are real
    if (col < 4) {
#pragma unroll
      for (int r2 = 0; r2 < 8; ++r2)
        oBuf[(base + r2 + 8 * g) * 4 + col] = oacc[r2];
    }
  }
}

// ---------------------------------------------------------------------------
// Kernel 3: per-ray transmittance scan (N=48) and final outputs
// out = [rgb (B*R*3)] ++ [depth (B*R)] ++ [wsum (B*R)] ++ [T (B*R)]
// ---------------------------------------------------------------------------
__global__ void render_kernel(const float* __restrict__ oBuf,
                              const float* __restrict__ jit,
                              float* __restrict__ out) {
  int ray = blockIdx.x * blockDim.x + threadIdx.x;  // ray = b*R + r
  const int BR = B_ * R_;
  if (ray >= BR) return;

  const float sdel = 1.f / (N_ - 1);
  size_t baseP = (size_t)ray * N_;
  float T = 1.f, rgb0 = 0.f, rgb1 = 0.f, rgb2 = 0.f, depth = 0.f, wsum = 0.f;
  float sd_cur = jit[baseP + 0] * sdel;

  for (int n = 0; n < N_; ++n) {
    float sd_next = (n < N_ - 1)
        ? (float)(n + 1) * sdel + jit[baseP + n + 1] * sdel : 0.f;
    float delta = (n < N_ - 1) ? (sd_next - sd_cur) : 1e10f;
    const float* oo = oBuf + (baseP + n) * 4;
    float dens  = softplus_f(oo[0]);
    // 1 - exp(-delta*dens) via bare v_exp_f32 (2^x with folded log2e)
    float alpha = 1.f - __builtin_amdgcn_exp2f(-delta * dens * LOG2E_F);
    float w = alpha * T;
    rgb0 += w * (sigmoid_f(oo[1]) * 1.002f - 0.001f);
    rgb1 += w * (sigmoid_f(oo[2]) * 1.002f - 0.001f);
    rgb2 += w * (sigmoid_f(oo[3]) * 1.002f - 0.001f);
    depth += w * sd_cur;
    wsum  += w;
    T *= (1.f - alpha + 1e-10f);
    sd_cur = sd_next;
  }
  out[(size_t)ray * 3 + 0] = rgb0;
  out[(size_t)ray * 3 + 1] = rgb1;
  out[(size_t)ray * 3 + 2] = rgb2;
  out[(size_t)BR * 3 + ray] = depth;
  out[(size_t)BR * 4 + ray] = wsum;
  out[(size_t)BR * 5 + ray] = T;
}

// ---------------------------------------------------------------------------
extern "C" void kernel_launch(void* const* d_in, const int* in_sizes, int n_in,
                              void* d_out, int out_size, void* d_ws,
                              size_t ws_size, hipStream_t stream) {
  const float* planes = (const float*)d_in[0];
  const float* ro     = (const float*)d_in[1];
  const float* rd     = (const float*)d_in[2];
  const float* jit    = (const float*)d_in[3];
  const float* W1     = (const float*)d_in[4];
  const float* b1     = (const float*)d_in[5];
  const float* W2     = (const float*)d_in[6];
  const float* b2     = (const float*)d_in[7];
  float* out = (float*)d_out;

  _Float16* xfeat = (_Float16*)d_ws;                               // 24 MB
  float* oBuf = (float*)((char*)d_ws + (size_t)BP * F_ * sizeof(_Float16));

  feat_kernel<<<(BP + 255) / 256, 256, 0, stream>>>(planes, ro, rd, jit, xfeat);
  mlp_kernel<<<512, 256, 0, stream>>>(xfeat, W1, b1, W2, b2, oBuf);
  render_kernel<<<(B_ * R_ + 255) / 256, 256, 0, stream>>>(oBuf, jit, out);
}
